// TrainGRPOModuleFusedLmHead_31138512896088
// MI455X (gfx1250) — matrix operand order repaired and checked by reading, not compiled, not run on hardware
//
#include <hip/hip_runtime.h>
#include <hip/hip_bf16.h>
#include <math.h>

// ---------------------------------------------------------------------------
// GRPO fused-LM-head loss for MI455X (gfx1250, wave32, WMMA bf16).
//   h:  [4,1024,2048] f32 (tokens 0..1022 used)   W: [2048,32768] f32
//   out: [1 + 4*1023] f32  = (loss, per_token_logps)
// Fast path: pre-convert h/W to bf16 (W transposed) -> conversion-free WMMA
// GEMM with ping-pong LDS. Fallback (small ws): convert-in-staging GEMM.
// ---------------------------------------------------------------------------

typedef __attribute__((ext_vector_type(16))) __bf16 v16bf;
typedef __attribute__((ext_vector_type(8)))  __bf16 v8bf;
typedef __attribute__((ext_vector_type(4)))  __bf16 v4bf;
typedef __attribute__((ext_vector_type(2)))  __bf16 v2bf;
typedef __attribute__((ext_vector_type(8)))  float  v8f;
typedef __attribute__((ext_vector_type(4)))  float  v4f;

#define B_DIM 4
#define S_DIM 1024
#define D_DIM 2048
#define V_DIM 32768
#define SM1   (S_DIM - 1)            // 1023
#define T_TOK (B_DIM * SM1)          // 4092 tokens
#define T_PAD 4096
#define TM 128                       // token tile
#define TN 128                       // vocab tile
#define KB 32                        // K step (WMMA bf16 K)
#define KP (KB + 8)                  // padded LDS row (80B: 16B-aligned)
#define NVT (V_DIM / TN)             // 256 vocab tiles
#define NTT ((T_TOK + TM - 1) / TM)  // 32 token tiles
#define NK  (D_DIM / KB)             // 64 K steps
#define EPS_LO 0.2f
#define EPS_HI 0.3f

__device__ __forceinline__ __bf16 f2bf(float f) {
  union { float f; unsigned u; } v; v.f = f;
  unsigned r = v.u + 0x7FFFu + ((v.u >> 16) & 1u);      // round-to-nearest-even
  unsigned short hs = (unsigned short)(r >> 16);
  union { unsigned short s; __bf16 b; } o; o.s = hs;
  return o.b;
}
__device__ __forceinline__ v2bf cvt2bf(float lo, float hi) {
#if __has_builtin(__builtin_amdgcn_cvt_pk_bf16_f32)
  return __builtin_amdgcn_cvt_pk_bf16_f32(lo, hi);
#else
  v2bf r; r[0] = f2bf(lo); r[1] = f2bf(hi); return r;
#endif
}

// ---------------------------------------------------------------------------
__global__ void grpo_init(float* accum) {
  accum[0] = 0.0f;   // masked loss numerator
  accum[1] = 0.0f;   // mask sum (denominator)
}

// ---------------------------------------------------------------------------
// Pre-pass A: h (f32, [:, :-1] remap + tail clamp) -> ht bf16 [T_PAD][D]
// ---------------------------------------------------------------------------
__global__ __launch_bounds__(256)
void grpo_conv_h(const float* __restrict__ hs, __bf16* __restrict__ ht) {
  const int t = blockIdx.x;                       // 0..T_PAD-1
  int tc = (t > T_TOK - 1) ? (T_TOK - 1) : t;     // clamp pad rows (benign)
  const int b = tc / SM1, s = tc % SM1;
  const size_t src = ((size_t)b * S_DIM + s) * D_DIM + (size_t)threadIdx.x * 8;
  v4f x0 = *(const v4f*)(hs + src);
  v4f x1 = *(const v4f*)(hs + src + 4);
  union { v8bf v; v2bf h[4]; } p;
  p.h[0] = cvt2bf(x0[0], x0[1]);  p.h[1] = cvt2bf(x0[2], x0[3]);
  p.h[2] = cvt2bf(x1[0], x1[1]);  p.h[3] = cvt2bf(x1[2], x1[3]);
  *(v8bf*)&ht[(size_t)t * D_DIM + (size_t)threadIdx.x * 8] = p.v;
}

// ---------------------------------------------------------------------------
// Pre-pass B: W f32 [D][V] -> Wt bf16 [V][D] (32x32 LDS tile transpose)
// ---------------------------------------------------------------------------
__global__ __launch_bounds__(256)
void grpo_conv_w(const float* __restrict__ W, __bf16* __restrict__ Wt) {
  __shared__ float tile[32][33];
  const int v0 = blockIdx.x * 32;
  const int d0 = blockIdx.y * 32;
  const int tx = threadIdx.x & 31;
  const int ty = threadIdx.x >> 5;                // 0..7
  #pragma unroll
  for (int i = 0; i < 4; ++i) {
    int d = ty * 4 + i;
    tile[d][tx] = W[(size_t)(d0 + d) * V_DIM + (size_t)(v0 + tx)];  // coalesced in v
  }
  __syncthreads();
  const int dp = threadIdx.x & 15;                // d-pair (2 bf16 per store)
  const int r0 = threadIdx.x >> 4;                // 0..15
  #pragma unroll
  for (int i = 0; i < 2; ++i) {
    int r = i * 16 + r0;                          // v row within tile
    v2bf p = cvt2bf(tile[dp * 2][r], tile[dp * 2 + 1][r]);
    *(v2bf*)&Wt[(size_t)(v0 + r) * D_DIM + (size_t)(d0 + dp * 2)] = p;  // coalesced in d
  }
}

// ---------------------------------------------------------------------------
// Shared tail: per-row softmax partials from the 16x128 accumulator strip.
// ---------------------------------------------------------------------------
__device__ __forceinline__ void emit_partials(const v8f* acc, int wave, int hf, int lr,
                                              int tt, int tv,
                                              float* __restrict__ pm,
                                              float* __restrict__ ps) {
  // C/D layout: VGPR j, lane<16 -> (M=j, N=lane); lane>=16 -> (M=j+8, N=lane-16).
  #pragma unroll
  for (int j = 0; j < 8; ++j) {
    float mx = acc[0][j];
    #pragma unroll
    for (int n = 1; n < 8; ++n) mx = fmaxf(mx, acc[n][j]);
    #pragma unroll
    for (int d = 1; d < 16; d <<= 1) mx = fmaxf(mx, __shfl_xor(mx, d, 32));
    float sm = 0.0f;
    #pragma unroll
    for (int n = 0; n < 8; ++n) sm += __expf(acc[n][j] - mx);
    #pragma unroll
    for (int d = 1; d < 16; d <<= 1) sm += __shfl_xor(sm, d, 32);
    if (lr == 0) {
      int t = tt * TM + wave * 16 + hf * 8 + j;
      if (t < T_TOK) {
        pm[(size_t)tv * T_PAD + t] = mx;
        ps[(size_t)tv * T_PAD + t] = sm;
      }
    }
  }
}

// ---------------------------------------------------------------------------
// FAST GEMM: bf16 inputs (ht row-major, Wt transposed) -> zero conversion
// VALU in the K loop. 256 threads = 8 waves; wave w owns rows 16w..16w+15.
// ---------------------------------------------------------------------------
__global__ __launch_bounds__(256)
void grpo_gemm_lse_bf16(const __bf16* __restrict__ ht,   // [T_PAD][D]
                        const __bf16* __restrict__ Wt,   // [V][D]
                        float* __restrict__ pm, float* __restrict__ ps)
{
  __shared__ __align__(16) __bf16 lA[2][TM][KP];   // [m][k]
  __shared__ __align__(16) __bf16 lB[2][TN][KP];   // [n][k]

  const int tid  = threadIdx.x;
  const int wave = tid >> 5;
  const int lane = tid & 31;
  const int hf   = lane >> 4;
  const int lr   = lane & 15;
  const int tv   = blockIdx.x;
  const int tt   = blockIdx.y;

  // staging: each thread moves one 16-bf16 half-row of A and of B per K step
  const int srow = tid >> 1;          // 0..127
  const int skh  = (tid & 1) * 16;    // k-half offset
  const __bf16* gA = ht + (size_t)(tt * TM + srow) * D_DIM + skh;
  const __bf16* gB = Wt + (size_t)(tv * TN + srow) * D_DIM + skh;

  v8bf ra0, ra1, rb0, rb1;
  auto load_tiles = [&](int k0) {
    ra0 = *(const v8bf*)(gA + k0);
    ra1 = *(const v8bf*)(gA + k0 + 8);
    rb0 = *(const v8bf*)(gB + k0);
    rb1 = *(const v8bf*)(gB + k0 + 8);
    if (k0 + KB < D_DIM) __builtin_prefetch(gB + k0 + KB, 0, 1);
  };
  auto store_tiles = [&](int buf) {
    *(v8bf*)&lA[buf][srow][skh]     = ra0;
    *(v8bf*)&lA[buf][srow][skh + 8] = ra1;
    *(v8bf*)&lB[buf][srow][skh]     = rb0;
    *(v8bf*)&lB[buf][srow][skh + 8] = rb1;
  };

  v8f acc[8] = { {}, {}, {}, {}, {}, {}, {}, {} };

  load_tiles(0);
  store_tiles(0);
  __syncthreads();

  for (int it = 0; it < NK; ++it) {
    const int buf = it & 1;
    const bool more = (it + 1 < NK);
    if (more) load_tiles((it + 1) * KB);

    union { v16bf v; v8bf h[2]; } a;   // A frag: slots 0-7 K=hf*8.., 8-15 +16
    const __bf16* pa = &lA[buf][wave * 16 + lr][hf * 8];
    a.h[0] = *(const v8bf*)(pa);
    a.h[1] = *(const v8bf*)(pa + 16);

    union { v16bf v; v8bf h[2]; } bfr[8];  // B frags: lane=N, slots 0-7 K=hf*8.., 8-15 +16
    #pragma unroll
    for (int n = 0; n < 8; ++n) {
      const __bf16* pb = &lB[buf][n * 16 + lr][hf * 8];
      bfr[n].h[0] = *(const v8bf*)(pb);
      bfr[n].h[1] = *(const v8bf*)(pb + 16);
    }
    #pragma unroll
    for (int n = 0; n < 8; ++n)
      acc[n] = __builtin_amdgcn_wmma_f32_16x16x32_bf16(
          false, a.v, false, bfr[n].v, (short)0, acc[n], false, false);

    if (more) store_tiles(buf ^ 1);
    __syncthreads();
  }

  emit_partials(acc, wave, hf, lr, tt, tv, pm, ps);
}

// ---------------------------------------------------------------------------
// FALLBACK GEMM: f32 inputs, convert during staging (used when ws is small).
// ---------------------------------------------------------------------------
__global__ __launch_bounds__(256)
void grpo_gemm_lse_conv(const float* __restrict__ hs,
                        const float* __restrict__ W,
                        float* __restrict__ pm, float* __restrict__ ps)
{
  __shared__ __align__(16) __bf16 lA[2][TM][KP];
  __shared__ __align__(16) __bf16 lB[2][TN][KP];

  const int tid  = threadIdx.x;
  const int wave = tid >> 5;
  const int lane = tid & 31;
  const int hf   = lane >> 4;
  const int lr   = lane & 15;
  const int tv   = blockIdx.x;
  const int tt   = blockIdx.y;

  const int am = tid >> 3;
  const int ak = (tid & 7) * 4;
  const int bn = (tid & 31) * 4;
  size_t abase[4];
  #pragma unroll
  for (int i = 0; i < 4; ++i) {
    int m = i * 32 + am;
    int t = tt * TM + m;
    if (t > T_TOK - 1) t = T_TOK - 1;
    int b = t / SM1, s = t % SM1;
    abase[i] = ((size_t)b * S_DIM + s) * (size_t)D_DIM;
  }

  v4f ra[4], rb[4];
  auto load_tiles = [&](int k0) {
    #pragma unroll
    for (int i = 0; i < 4; ++i)
      ra[i] = *(const v4f*)(hs + abase[i] + (size_t)(k0 + ak));
    #pragma unroll
    for (int i = 0; i < 4; ++i) {
      int k = i * 8 + wave;
      rb[i] = *(const v4f*)(W + (size_t)(k0 + k) * V_DIM + (size_t)(tv * TN + bn));
    }
  };
  auto store_tiles = [&](int buf) {
    #pragma unroll
    for (int i = 0; i < 4; ++i) {
      int m = i * 32 + am;
      union { v4bf v; v2bf h[2]; } p;
      p.h[0] = cvt2bf(ra[i][0], ra[i][1]);
      p.h[1] = cvt2bf(ra[i][2], ra[i][3]);
      *(v4bf*)&lA[buf][m][ak] = p.v;
    }
    #pragma unroll
    for (int i = 0; i < 4; ++i) {
      int k = i * 8 + wave;
      #pragma unroll
      for (int j = 0; j < 4; j += 2) {
        v2bf p = cvt2bf(rb[i][j], rb[i][j + 1]);
        lB[buf][bn + j][k]     = p[0];
        lB[buf][bn + j + 1][k] = p[1];
      }
    }
  };

  v8f acc[8] = { {}, {}, {}, {}, {}, {}, {}, {} };

  load_tiles(0);
  store_tiles(0);
  __syncthreads();

  for (int it = 0; it < NK; ++it) {
    const int buf = it & 1;
    const bool more = (it + 1 < NK);
    if (more) load_tiles((it + 1) * KB);

    union { v16bf v; v8bf h[2]; } a;
    const __bf16* pa = &lA[buf][wave * 16 + lr][hf * 8];
    a.h[0] = *(const v8bf*)(pa);
    a.h[1] = *(const v8bf*)(pa + 16);

    union { v16bf v; v8bf h[2]; } bfr[8];
    #pragma unroll
    for (int n = 0; n < 8; ++n) {
      const __bf16* pb = &lB[buf][n * 16 + lr][hf * 8];
      bfr[n].h[0] = *(const v8bf*)(pb);
      bfr[n].h[1] = *(const v8bf*)(pb + 16);
    }
    #pragma unroll
    for (int n = 0; n < 8; ++n)
      acc[n] = __builtin_amdgcn_wmma_f32_16x16x32_bf16(
          false, a.v, false, bfr[n].v, (short)0, acc[n], false, false);

    if (more) store_tiles(buf ^ 1);
    __syncthreads();
  }

  emit_partials(acc, wave, hf, lr, tt, tv, pm, ps);
}

// ---------------------------------------------------------------------------
// Per-token streaming-LSE merge + chosen logit + PPO-clip loss (1 wave/token).
// ---------------------------------------------------------------------------
__global__ __launch_bounds__(256)
void grpo_reduce_loss(const float* __restrict__ hs, const float* __restrict__ W,
                      const int* __restrict__ ids, const int* __restrict__ labels,
                      const float* __restrict__ adv, const float* __restrict__ oldlp,
                      const float* __restrict__ pm, const float* __restrict__ ps,
                      float* __restrict__ out, float* __restrict__ accum)
{
  const int wave = threadIdx.x >> 5;
  const int lane = threadIdx.x & 31;
  const int t = blockIdx.x * 8 + wave;
  if (t >= T_TOK) return;
  const int b = t / SM1, s = t % SM1;

  float mx = -INFINITY;
  for (int j = lane; j < NVT; j += 32) mx = fmaxf(mx, pm[(size_t)j * T_PAD + t]);
  #pragma unroll
  for (int d = 1; d < 32; d <<= 1) mx = fmaxf(mx, __shfl_xor(mx, d, 32));

  float sm = 0.0f;
  for (int j = lane; j < NVT; j += 32)
    sm += ps[(size_t)j * T_PAD + t] * __expf(pm[(size_t)j * T_PAD + t] - mx);
  #pragma unroll
  for (int d = 1; d < 32; d <<= 1) sm += __shfl_xor(sm, d, 32);
  const float lse = mx + __logf(sm);

  const int c = ids[b * S_DIM + s + 1];
  float dot = 0.0f;
  for (int d = lane; d < D_DIM; d += 32)
    dot += hs[((size_t)b * S_DIM + s) * D_DIM + d] * W[(size_t)d * V_DIM + c];
  #pragma unroll
  for (int d = 1; d < 32; d <<= 1) dot += __shfl_xor(dot, d, 32);

  if (lane == 0) {
    const float logp = dot - lse;            // TEMP == 1.0
    out[1 + t] = logp;
    const float ratio = __expf(logp - oldlp[b * SM1 + s]);
    const float a  = adv[b];
    const float l1 = ratio * a;
    const float l2 = fminf(fmaxf(ratio, 1.0f - EPS_LO), 1.0f + EPS_HI) * a;
    const float ptl = -fminf(l1, l2);
    const float mk  = (float)labels[b * S_DIM + s + 1];
    atomicAdd(&accum[0], ptl * mk);
    atomicAdd(&accum[1], mk);
  }
}

// ---------------------------------------------------------------------------
__global__ void grpo_finalize(const float* __restrict__ accum, float* __restrict__ out) {
  out[0] = accum[0] / accum[1];
}

// ---------------------------------------------------------------------------
extern "C" void kernel_launch(void* const* d_in, const int* in_sizes, int n_in,
                              void* d_out, int out_size, void* d_ws, size_t ws_size,
                              hipStream_t stream) {
  const float* hs     = (const float*)d_in[0];
  const float* W      = (const float*)d_in[1];
  const int*   ids    = (const int*)  d_in[2];
  const int*   labels = (const int*)  d_in[3];
  const float* adv    = (const float*)d_in[4];
  const float* oldlp  = (const float*)d_in[5];
  float* out = (float*)d_out;

  char* ws = (char*)d_ws;
  float* accum = (float*)ws;                                  // 2 floats
  float* pm = (float*)(ws + 256);                             // [NVT][T_PAD]
  float* ps = pm + (size_t)NVT * T_PAD;                       // [NVT][T_PAD]
  __bf16* ht = (__bf16*)(ps + (size_t)NVT * T_PAD);           // [T_PAD][D]
  __bf16* Wt = ht + (size_t)T_PAD * D_DIM;                    // [V][D]

  const size_t need_fast = 256
      + (size_t)2 * NVT * T_PAD * sizeof(float)
      + (size_t)T_PAD * D_DIM * sizeof(__bf16)
      + (size_t)V_DIM * D_DIM * sizeof(__bf16);

  grpo_init<<<1, 1, 0, stream>>>(accum);
  if (ws_size >= need_fast) {
    grpo_conv_h<<<dim3(T_PAD), 256, 0, stream>>>(hs, ht);
    grpo_conv_w<<<dim3(V_DIM / 32, D_DIM / 32), 256, 0, stream>>>(W, Wt);
    grpo_gemm_lse_bf16<<<dim3(NVT, NTT), 256, 0, stream>>>(ht, Wt, pm, ps);
  } else {
    grpo_gemm_lse_conv<<<dim3(NVT, NTT), 256, 0, stream>>>(hs, W, pm, ps);
  }
  grpo_reduce_loss<<<dim3((T_TOK + 7) / 8), 256, 0, stream>>>(
      hs, W, ids, labels, adv, oldlp, pm, ps, out, accum);
  grpo_finalize<<<1, 1, 0, stream>>>(accum, out);
}
